// EncoderAttention_27573690040685
// MI455X (gfx1250) — matrix operand order repaired and checked by reading, not compiled, not run on hardware
//
#include <hip/hip_runtime.h>
#include <hip/hip_bf16.h>

// ---------------------------------------------------------------------------
// EncoderAttention on gfx1250 (MI455X): f16 WMMA pipeline with f32 accumulate.
//   K0: convert x + weights fp32->f16
//   K1: per-head GEMM (X @ W^T), fused RoPE + scale in f32 accumulators
//       (V pass stores transposed per head for fragment-friendly PV loads)
//   K2: flash attention; 8 waves/block share one (b,head); 64-key K/V chunks
//       staged into LDS via GLOBAL_LOAD_ASYNC_TO_LDS (ASYNCcnt path);
//       row-sums via P @ ones WMMA; P transposed via per-wave LDS tile
//   K3: out-projection GEMM, f32 out
// ---------------------------------------------------------------------------

constexpr int BATCH = 8;
constexpr int SEQ   = 2048;
constexpr int HID   = 384;
constexpr int NHEAD = 6;
constexpr int HDIM  = 64;
constexpr int BSROWS = BATCH * SEQ;                 // 16384
constexpr size_t BSH = (size_t)BSROWS * HID;        // 6,291,456 elems
constexpr size_t HH  = (size_t)HID * HID;           // 147,456 elems

typedef __attribute__((ext_vector_type(16))) _Float16 v16h;
typedef __attribute__((ext_vector_type(8)))  _Float16 v8h;
typedef __attribute__((ext_vector_type(8)))  float    v8f;
typedef __attribute__((ext_vector_type(4)))  int      v4i;

#if __has_builtin(__builtin_amdgcn_global_load_async_to_lds_b128) && \
    __has_builtin(__builtin_amdgcn_s_wait_asynccnt)
#define USE_ASYNC_LDS 1
#else
#define USE_ASYNC_LDS 0
#endif

#define AS_GLOBAL __attribute__((address_space(1)))
#define AS_LDS    __attribute__((address_space(3)))

// copy 16 bytes global -> LDS (async path avoids the VGPR round-trip)
__device__ __forceinline__ void copy16_g2l(const _Float16* gsrc,
                                           _Float16* ldst) {
#if USE_ASYNC_LDS
    AS_GLOBAL v4i* g =
        (AS_GLOBAL v4i*)(v4i*)(void*)const_cast<_Float16*>(gsrc);
    AS_LDS v4i* l = (AS_LDS v4i*)(v4i*)(void*)ldst;
    __builtin_amdgcn_global_load_async_to_lds_b128(g, l, 0, 0);
#else
    *reinterpret_cast<v8h*>(ldst) = *reinterpret_cast<const v8h*>(gsrc);
#endif
}

__device__ __forceinline__ void wait_g2l() {
#if USE_ASYNC_LDS
    __builtin_amdgcn_s_wait_asynccnt(0);
#endif
}

// B-fragment: lane holds one column; halves 0..7 at p, halves 8..15 at p+16
__device__ __forceinline__ v16h ld_frag_contig(const _Float16* p) {
    v8h lo = *reinterpret_cast<const v8h*>(p);
    v8h hi = *reinterpret_cast<const v8h*>(p + 16);
    v16h r;
#pragma unroll
    for (int i = 0; i < 8; ++i) { r[i] = lo[i]; r[i + 8] = hi[i]; }
    return r;
}

// A-fragment: halves 0..7 at p, halves 8..15 at p+16 (K = (h/8)*16 + g*8 + h%8)
__device__ __forceinline__ v16h ld_frag_split16(const _Float16* p) {
    v8h lo = *reinterpret_cast<const v8h*>(p);
    v8h hi = *reinterpret_cast<const v8h*>(p + 16);
    v16h r;
#pragma unroll
    for (int i = 0; i < 8; ++i) { r[i] = lo[i]; r[i + 8] = hi[i]; }
    return r;
}

// ---------------------------------------------------------------------------
// K0: fp32 -> f16 conversion of x and the 4 weight matrices
// ---------------------------------------------------------------------------
__global__ void convert_inputs_kernel(const float* __restrict__ x,
                                      const float* __restrict__ wq,
                                      const float* __restrict__ wk,
                                      const float* __restrict__ wv,
                                      const float* __restrict__ wo,
                                      _Float16* __restrict__ ws) {
    size_t i = (size_t)blockIdx.x * blockDim.x + threadIdx.x;
    constexpr size_t total = BSH + 4 * HH;
    if (i >= total) return;
    float v;
    if (i < BSH) {
        v = x[i];
    } else {
        size_t j = i - BSH;
        int w = (int)(j / HH);
        size_t k = j % HH;
        const float* src = (w == 0) ? wq : (w == 1) ? wk : (w == 2) ? wv : wo;
        v = src[k];
    }
    ws[i] = (_Float16)v;
}

// ---------------------------------------------------------------------------
// K1/K3: per-head GEMM  C[16 x 64] = scale * (X[16 x 384] @ W^T)
//   DO_ROPE: apply rotary embedding in the f32 accumulators
//   OUT_MODE: 0 = f16 row-major [bs, H]; 1 = f16 transposed [b,h,d,s];
//             2 = f32 row-major (final output)
// ---------------------------------------------------------------------------
template <int DO_ROPE, int OUT_MODE>
__global__ __launch_bounds__(256)
void gemm_head_kernel(const _Float16* __restrict__ X,
                      const _Float16* __restrict__ W,
                      const float* __restrict__ ctab,
                      const float* __restrict__ stab,
                      float out_scale,
                      _Float16* __restrict__ out_h,
                      float* __restrict__ out_f) {
    const int lane = threadIdx.x & 31;
    const int wid  = (int)((blockIdx.x * blockDim.x + threadIdx.x) >> 5);
    const int qtile = wid / NHEAD;
    const int head  = wid % NHEAD;
    const int row0  = qtile * 16;
    const int g  = lane >> 4;     // half-wave group (0/1)
    const int ln = lane & 15;

    v8f acc[4] = {};

    for (int kk = 0; kk < HID; kk += 32) {
        // A fragment: rows of X (lane = M), two contiguous 8-half chunks
        const _Float16* pa = X + (size_t)(row0 + ln) * HID + kk + g * 8;
        v16h a = ld_frag_split16(pa);
#pragma unroll
        for (int t = 0; t < 4; ++t) {
            // B = W^T: lane ln is output column n -> row (head*64+t*16+ln) of W
            const _Float16* pb =
                W + (size_t)(head * HDIM + t * 16 + ln) * HID + kk + g * 8;
            v16h b = ld_frag_contig(pb);
            acc[t] = __builtin_amdgcn_wmma_f32_16x16x32_f16(
                false, a, false, b, (short)0, acc[t], false, false);
        }
    }

    if constexpr (DO_ROPE) {
        // col d in tile t is paired with d+32 in tile t+2, same lane/element
#pragma unroll
        for (int t = 0; t < 2; ++t) {
#pragma unroll
            for (int e = 0; e < 8; ++e) {
                int m = e + 8 * g;
                int spos = (row0 + m) & (SEQ - 1);
                int d2 = t * 16 + ln;                  // 0..31
                float c  = ctab[(size_t)spos * (HDIM / 2) + d2];
                float sn = stab[(size_t)spos * (HDIM / 2) + d2];
                float lo = acc[t][e];
                float hi = acc[t + 2][e];
                acc[t][e]     = lo * c - hi * sn;
                acc[t + 2][e] = hi * c + lo * sn;
            }
        }
    }

#pragma unroll
    for (int t = 0; t < 4; ++t) {
#pragma unroll
        for (int e = 0; e < 8; ++e) {
            float v = acc[t][e] * out_scale;
            int m = e + 8 * g;
            int row = row0 + m;
            if constexpr (OUT_MODE == 0) {
                out_h[(size_t)row * HID + head * HDIM + t * 16 + ln] =
                    (_Float16)v;
            } else if constexpr (OUT_MODE == 1) {
                int b = row / SEQ;
                int spos = row & (SEQ - 1);
                int d = t * 16 + ln;
                out_h[((size_t)(b * NHEAD + head) * HDIM + d) * SEQ + spos] =
                    (_Float16)v;
            } else {
                out_f[(size_t)row * HID + head * HDIM + t * 16 + ln] = v;
            }
        }
    }
}

// ---------------------------------------------------------------------------
// K2: flash attention. One block = (b, head, 128 query rows); 8 waves each own
// a 16-row tile and share 64-key K/V chunks staged through LDS.
// ---------------------------------------------------------------------------
constexpr int WAVES_PER_BLOCK = 8;
constexpr int KVCHUNK = 64;

__global__ __launch_bounds__(256)
void attention_kernel(const _Float16* __restrict__ Q,
                      const _Float16* __restrict__ K,
                      const _Float16* __restrict__ Vt,
                      const float* __restrict__ maskp,
                      _Float16* __restrict__ Ah) {
    __shared__ __align__(16) _Float16 s_k[KVCHUNK][HDIM];      // [kv][d]  8KB
    __shared__ __align__(16) _Float16 s_v[HDIM][KVCHUNK];      // [d][kv]  8KB
    __shared__ __align__(16) _Float16 pbuf[WAVES_PER_BLOCK][16 * KVCHUNK]; // 16KB

    const int tid  = threadIdx.x;
    const int lane = tid & 31;
    const int wib  = tid >> 5;
    // block -> (b, head, qgroup)
    constexpr int QGROUPS = SEQ / (16 * WAVES_PER_BLOCK);      // 16
    const int qgroup = blockIdx.x % QGROUPS;
    const int head   = (blockIdx.x / QGROUPS) % NHEAD;
    const int b      = blockIdx.x / (QGROUPS * NHEAD);
    const int s0     = qgroup * 16 * WAVES_PER_BLOCK + wib * 16;
    const int row0   = b * SEQ + s0;
    const int g  = lane >> 4;
    const int ln = lane & 15;

    const _Float16* Kbase  = K + (size_t)b * SEQ * HID + head * HDIM;
    const _Float16* Vtbase = Vt + (size_t)(b * NHEAD + head) * HDIM * SEQ;

    // cooperative-copy slices: 64x64 halves = 8KB each; 32B per thread
    const int crow = tid >> 2;             // 0..63
    const int ccol = (tid & 3) * 16;       // 0,16,32,48 (halves)

    // Q A-fragments for both 32-wide d-chunks (Q is pre-scaled by 1/8)
    v16h aq[2];
#pragma unroll
    for (int c = 0; c < 2; ++c) {
        const _Float16* pa =
            Q + (size_t)(row0 + ln) * HID + head * HDIM + c * 32 + g * 8;
        aq[c] = ld_frag_split16(pa);
    }

    // ones B-fragment for row-sum WMMA
    v16h bones;
#pragma unroll
    for (int i = 0; i < 16; ++i) bones[i] = (_Float16)1.0f;

    float mrow[8], lrow[8];
    v8f oacc[4] = {};
#pragma unroll
    for (int e = 0; e < 8; ++e) { mrow[e] = -3.0e38f; lrow[e] = 0.0f; }

    _Float16* pb = &pbuf[wib][0];

    for (int kv0 = 0; kv0 < SEQ; kv0 += KVCHUNK) {
        // ---- stage K (64x64, [kv][d]) and V^T (64x64, [d][kv]) into LDS
        {
            const _Float16* ksrc = Kbase + (size_t)(kv0 + crow) * HID + ccol;
            copy16_g2l(ksrc,     &s_k[crow][ccol]);
            copy16_g2l(ksrc + 8, &s_k[crow][ccol + 8]);
            const _Float16* vsrc = Vtbase + (size_t)crow * SEQ + kv0 + ccol;
            copy16_g2l(vsrc,     &s_v[crow][ccol]);
            copy16_g2l(vsrc + 8, &s_v[crow][ccol + 8]);
            if (kv0 + KVCHUNK < SEQ) {   // pull next chunk toward the caches
                __builtin_prefetch(ksrc + (size_t)KVCHUNK * HID, 0, 1);
                __builtin_prefetch(vsrc + KVCHUNK, 0, 1);
            }
        }
        wait_g2l();
        __syncthreads();

        // ---- S = Q @ K^T for 4 kv sub-tiles (cols) x 2 d-chunks (K dim)
        v8f sacc[4] = {};
#pragma unroll
        for (int sub = 0; sub < 4; ++sub) {
#pragma unroll
            for (int c = 0; c < 2; ++c) {
                v16h bk = ld_frag_contig(&s_k[sub * 16 + ln][c * 32 + g * 8]);
                sacc[sub] = __builtin_amdgcn_wmma_f32_16x16x32_f16(
                    false, aq[c], false, bk, (short)0, sacc[sub], false, false);
            }
        }

        // ---- additive mask (softmax scale already folded into Q)
#pragma unroll
        for (int sub = 0; sub < 4; ++sub) {
            float mk = maskp[(size_t)b * SEQ + kv0 + sub * 16 + ln];
#pragma unroll
            for (int e = 0; e < 8; ++e) sacc[sub][e] += mk;
        }

        // ---- online softmax: row max via 16-lane shuffle tree (once/chunk)
        float corr[8];
#pragma unroll
        for (int e = 0; e < 8; ++e) {
            float rm = fmaxf(fmaxf(sacc[0][e], sacc[1][e]),
                             fmaxf(sacc[2][e], sacc[3][e]));
#pragma unroll
            for (int off = 8; off >= 1; off >>= 1)
                rm = fmaxf(rm, __shfl_xor(rm, off, 32));
            float mnew = fmaxf(mrow[e], rm);
            corr[e] = __expf(mrow[e] - mnew);
            mrow[e] = mnew;
        }

        // ---- P = exp(S - m), streamed straight into the per-wave LDS tile
        // (C-layout -> A-layout transpose; LDS ops are wave-ordered)
#pragma unroll
        for (int sub = 0; sub < 4; ++sub) {
#pragma unroll
            for (int e = 0; e < 8; ++e) {
                float p = __expf(sacc[sub][e] - mrow[e]);
                pb[(e + 8 * g) * KVCHUNK + sub * 16 + ln] = (_Float16)p;
            }
        }
#pragma unroll
        for (int t = 0; t < 4; ++t)
#pragma unroll
            for (int e = 0; e < 8; ++e) oacc[t][e] *= corr[e];

        asm volatile("" ::: "memory");  // keep compiler from reordering ds ops
        v16h pa0 = ld_frag_split16(pb + ln * KVCHUNK + g * 8);        // K 0..31
        v16h pa1 = ld_frag_split16(pb + ln * KVCHUNK + 32 + g * 8);   // K 32..63

        // ---- row sums via P @ ones (replicated across columns, C-layout)
        v8f sumt = {};
        sumt = __builtin_amdgcn_wmma_f32_16x16x32_f16(
            false, pa0, false, bones, (short)0, sumt, false, false);
        sumt = __builtin_amdgcn_wmma_f32_16x16x32_f16(
            false, pa1, false, bones, (short)0, sumt, false, false);
#pragma unroll
        for (int e = 0; e < 8; ++e) lrow[e] = lrow[e] * corr[e] + sumt[e];

        // ---- O += P @ V  (s_v is [d][kv] so B-fragments are contiguous)
#pragma unroll
        for (int t = 0; t < 4; ++t) {
            v16h bv0 = ld_frag_contig(&s_v[t * 16 + ln][g * 8]);
            oacc[t] = __builtin_amdgcn_wmma_f32_16x16x32_f16(
                false, pa0, false, bv0, (short)0, oacc[t], false, false);
            v16h bv1 = ld_frag_contig(&s_v[t * 16 + ln][32 + g * 8]);
            oacc[t] = __builtin_amdgcn_wmma_f32_16x16x32_f16(
                false, pa1, false, bv1, (short)0, oacc[t], false, false);
        }
        __syncthreads();   // protect s_k/s_v before next chunk overwrites
    }

    // ---- normalize and store attention output (row-major f16)
#pragma unroll
    for (int e = 0; e < 8; ++e) {
        float inv = 1.0f / lrow[e];
        int m = e + 8 * g;
        int row = row0 + m;
#pragma unroll
        for (int t = 0; t < 4; ++t) {
            Ah[(size_t)row * HID + head * HDIM + t * 16 + ln] =
                (_Float16)(oacc[t][e] * inv);
        }
    }
}

// ---------------------------------------------------------------------------
// Host-side launch
// ---------------------------------------------------------------------------
extern "C" void kernel_launch(void* const* d_in, const int* in_sizes, int n_in,
                              void* d_out, int out_size, void* d_ws,
                              size_t ws_size, hipStream_t stream) {
    const float* x    = (const float*)d_in[0];
    const float* ctab = (const float*)d_in[1];
    const float* stab = (const float*)d_in[2];
    const float* mask = (const float*)d_in[3];
    const float* wq   = (const float*)d_in[4];
    const float* wk   = (const float*)d_in[5];
    const float* wv   = (const float*)d_in[6];
    const float* wo   = (const float*)d_in[7];
    float* out = (float*)d_out;

    _Float16* ws  = (_Float16*)d_ws;
    _Float16* Xh  = ws;                 // [BS, H]
    _Float16* Wqh = Xh + BSH;           // [H, H]
    _Float16* Wkh = Wqh + HH;
    _Float16* Wvh = Wkh + HH;
    _Float16* Woh = Wvh + HH;
    _Float16* Qh  = Woh + HH;           // [BS, H] post-RoPE, pre-scaled 1/8
    _Float16* Kh  = Qh + BSH;           // [BS, H] post-RoPE
    _Float16* Vt  = Kh + BSH;           // [B, NH, HD, S] transposed
    _Float16* Ah  = Vt + BSH;           // [BS, H] attention output

    {
        size_t total = BSH + 4 * HH;
        int blocks = (int)((total + 255) / 256);
        convert_inputs_kernel<<<blocks, 256, 0, stream>>>(x, wq, wk, wv, wo, ws);
    }

    const int gblocks = (BSROWS / 16) * NHEAD / WAVES_PER_BLOCK;  // 768

    // Q projection: fused RoPE + 1/sqrt(HD) softmax scale folded in
    gemm_head_kernel<1, 0><<<gblocks, 256, 0, stream>>>(Xh, Wqh, ctab, stab,
                                                        0.125f, Qh, nullptr);
    // K projection with fused RoPE
    gemm_head_kernel<1, 0><<<gblocks, 256, 0, stream>>>(Xh, Wkh, ctab, stab,
                                                        1.0f, Kh, nullptr);
    // V projection, stored transposed per head
    gemm_head_kernel<0, 1><<<gblocks, 256, 0, stream>>>(Xh, Wvh, nullptr, nullptr,
                                                        1.0f, Vt, nullptr);
    // Flash attention (768 blocks: 8 batches x 6 heads x 16 q-groups)
    attention_kernel<<<BATCH * NHEAD * (SEQ / (16 * WAVES_PER_BLOCK)),
                       256, 0, stream>>>(Qh, Kh, Vt, mask, Ah);
    // Output projection, f32 straight to d_out
    gemm_head_kernel<0, 2><<<gblocks, 256, 0, stream>>>(Ah, Woh, nullptr, nullptr,
                                                        1.0f, nullptr, out);
}
